// fbank_layer_81784767251215
// MI455X (gfx1250) — matrix operand order repaired and checked by reading, not compiled, not run on hardware
//
#include <hip/hip_runtime.h>

typedef __attribute__((ext_vector_type(16))) _Float16 v16h;
typedef __attribute__((ext_vector_type(8)))  _Float16 v8h;
typedef __attribute__((ext_vector_type(8)))  float    v8f;

#define NFFT       512
#define NBINS      257          // NFFT/2 + 1
#define NFILT      64
#define FRAME_LEN  400
#define FRAME_STEP 160
#define NUM_FRAMES 50000

#define KPAD   416              // FRAME_LEN padded to 13*32
#define KSTEPS 13
#define NPAD   528              // 2*NBINS = 514 padded to 33*16
#define NTILES 33
#define MELK   288              // NBINS padded to 9*32
#define MELKST 9

#define MROWS  32               // frames per workgroup (2 M-tiles)
#define A_STRIDE 424            // f16 elems; 212 dwords -> bank rotation 20
#define P_STRIDE 296            // f16 elems; 148 dwords -> bank rotation 20

// ---------------------------------------------------------------------------
// Transposed DFT basis Bt[NPAD][KPAD] f16. Column n = 2*bin + (0:cos,1:sin).
// Integer phase reduction keeps sincos args in [0, 2pi) (exact periodicity).
// ---------------------------------------------------------------------------
__global__ void build_basis_kernel(_Float16* __restrict__ Bt) {
    int idx = blockIdx.x * blockDim.x + threadIdx.x;
    if (idx >= NPAD * KPAD) return;
    int n = idx / KPAD;
    int k = idx - n * KPAD;
    float v = 0.0f;
    if (n < 2 * NBINS && k < FRAME_LEN) {
        int bin = n >> 1;
        int ph  = (bin * k) & (NFFT - 1);
        float th = (float)ph * (6.283185307179586f / (float)NFFT);
        v = (n & 1) ? __sinf(th) : __cosf(th);
    }
    Bt[idx] = (_Float16)v;
}

// ---------------------------------------------------------------------------
// Transposed, K-padded mel filters Ft[NFILT][MELK] f16.
// Input `filters` is (NBINS, NFILT) row-major f32.
// ---------------------------------------------------------------------------
__global__ void build_filt_kernel(const float* __restrict__ filters,
                                  _Float16* __restrict__ Ft) {
    int idx = blockIdx.x * blockDim.x + threadIdx.x;
    if (idx >= NFILT * MELK) return;
    int n = idx / MELK;
    int k = idx - n * MELK;
    float v = (k < NBINS) ? filters[k * NFILT + n] : 0.0f;
    Ft[idx] = (_Float16)v;
}

// ---------------------------------------------------------------------------
// Fused kernel: 1 workgroup (8 wave32) = 32 frames (2 M-tiles).
//   Phase 0: pre-emphasis + frame gather -> LDS A (f16); zero mel-K pad
//   Phase 1: DFT GEMM (each B tile reused for 2 M-tiles); re^2+im^2 paired
//            cross-lane via shfl_xor, f16 power straight into ldsP
//   Phase 2: mel GEMM, 8 wave-tiles (2 Mtiles x 4 Ntiles) via WMMA
//   Phase 3: per-frame mean / population-std
//   Phase 4: normalize + store f32 (guarded for frame overrun)
// ---------------------------------------------------------------------------
__launch_bounds__(256)
__global__ void fbank_main_kernel(const float* __restrict__ x,
                                  const _Float16* __restrict__ Bt,
                                  const _Float16* __restrict__ Ft,
                                  float* __restrict__ out) {
    __shared__ _Float16 ldsA[MROWS * A_STRIDE];   // 27,136 B
    __shared__ _Float16 ldsP[MROWS * P_STRIDE];   // 18,944 B
    __shared__ float    ldsF[MROWS * NFILT];      //  8,192 B
    __shared__ float    ldsM[MROWS];
    __shared__ float    ldsS[MROWS];

    const int tid   = threadIdx.x;
    const int wave  = tid >> 5;          // 8 wave32 per block
    const int lane  = tid & 31;
    const int l16   = lane & 15;
    const int lhalf = lane >> 4;         // 0 or 1
    const int f0    = blockIdx.x * MROWS;

    // ---- Phase 0: pre-emphasis + windowing into LDS (f16) ----
    // (Out-of-range frames still address valid audio; stores are guarded later.)
    for (int r = 0; r < MROWS; ++r) {
        int frame = f0 + r;
        if (frame >= NUM_FRAMES) frame = NUM_FRAMES - 1;   // clamp, keeps loads safe
        const int base = frame * FRAME_STEP;
        for (int k = tid; k < KPAD; k += 256) {
            float v = 0.0f;
            if (k < FRAME_LEN) {
                int i = base + k;
                float xi = x[i];
                float xm = (i > 0) ? x[i - 1] : 0.0f;
                v = xi - 0.97f * xm;             // y[0] = x[0]
            }
            ldsA[r * A_STRIDE + k] = (_Float16)v;
        }
    }
    // zero-fill mel-K pad columns (bins 264..287) of ldsP
    for (int i = tid; i < MROWS * (MELK - NPAD / 2); i += 256) {
        const int r = i / (MELK - NPAD / 2);
        const int c = (NPAD / 2) + (i - r * (MELK - NPAD / 2));
        ldsP[r * P_STRIDE + c] = (_Float16)0.0f;
    }
    __syncthreads();

    // ---- Phase 1: DFT GEMM. Each B register tile feeds 2 M-tiles. ----
    // A lane layout: row M = l16 (+16 for the 2nd M-tile); K chunks
    // {0..15} (lanes 0-15) or {16..31} (lanes 16-31).
    // B lane layout: column N = n0+l16, 16 contiguous K starting at lhalf*16.
    {
        const _Float16* aRow0 = ldsA + l16 * A_STRIDE + lhalf * 16;
        const _Float16* aRow1 = aRow0 + 16 * A_STRIDE;
        const float sc = 1.0f / (float)NFFT;
        for (int it = 0; it < 5; ++it) {                 // 8 waves * 5 >= 33 tiles
            const int t = wave + it * 8;
            if (t < NTILES) {
                const int n0 = t * 16;
                const _Float16* bptr = Bt + (n0 + l16) * KPAD + lhalf * 16;
                v8f acc0 = {};
                v8f acc1 = {};
                for (int kk = 0; kk < KSTEPS; ++kk) {
                    union { v16h v; v8h h[2]; } a0, a1, b;
                    b.v     = *(const v16h*)(bptr + kk * 32);
                    a0.h[0] = *(const v8h*)(aRow0 + kk * 32);
                    a0.h[1] = *(const v8h*)(aRow0 + kk * 32 + 8);
                    a1.h[0] = *(const v8h*)(aRow1 + kk * 32);
                    a1.h[1] = *(const v8h*)(aRow1 + kk * 32 + 8);
                    if (kk + 1 < KSTEPS)
                        __builtin_prefetch(bptr + (kk + 1) * 32, 0, 3);
                    acc0 = __builtin_amdgcn_wmma_f32_16x16x32_f16(
                        false, a0.v, false, b.v, (short)0, acc0, false, false);
                    acc1 = __builtin_amdgcn_wmma_f32_16x16x32_f16(
                        false, a1.v, false, b.v, (short)0, acc1, false, false);
                }
                // D layout: VGPR d, lanes 0-15 -> (M=d, N=n0+l16); 16-31 -> M=d+8.
                // re/im of bin (n0+l16)>>1 sit in adjacent lanes: pair via shfl_xor.
                #pragma unroll
                for (int d = 0; d < 8; ++d) {
                    const int row = d + lhalf * 8;
                    float p0 = acc0[d] * acc0[d];
                    float p1 = acc1[d] * acc1[d];
                    const float q0 = __shfl_xor(p0, 1, 32);
                    const float q1 = __shfl_xor(p1, 1, 32);
                    if ((l16 & 1) == 0) {
                        const int bin = (n0 + l16) >> 1;
                        ldsP[row * P_STRIDE + bin]        = (_Float16)((p0 + q0) * sc);
                        ldsP[(row + 16) * P_STRIDE + bin] = (_Float16)((p1 + q1) * sc);
                    }
                }
            }
        }
    }
    __syncthreads();

    // ---- Phase 2: mel GEMM, feat(32x64) = P(32xMELK) * Filt(MELKx64) ----
    // 8 wave-tiles: wave = mtile*4 + ntile
    {
        const int mtile = wave >> 2;
        const int n0    = (wave & 3) * 16;
        const _Float16* pRow = ldsP + (mtile * 16 + l16) * P_STRIDE + lhalf * 16;
        const _Float16* fptr = Ft + (n0 + l16) * MELK + lhalf * 16;
        v8f acc = {};
        for (int kk = 0; kk < MELKST; ++kk) {
            union { v16h v; v8h h[2]; } a, b;
            a.h[0] = *(const v8h*)(pRow + kk * 32);
            a.h[1] = *(const v8h*)(pRow + kk * 32 + 8);
            b.v    = *(const v16h*)(fptr + kk * 32);
            acc = __builtin_amdgcn_wmma_f32_16x16x32_f16(
                false, a.v, false, b.v, (short)0, acc, false, false);
        }
        #pragma unroll
        for (int d = 0; d < 8; ++d) {
            const int row = mtile * 16 + d + lhalf * 8;
            ldsF[row * NFILT + n0 + l16] = acc[d] + 1e-30f;
        }
    }
    __syncthreads();

    // ---- Phase 3: per-frame mean / std over 64 mel bins ----
    if (tid < MROWS) {
        float s = 0.0f, s2 = 0.0f;
        #pragma unroll 8
        for (int n = 0; n < NFILT; ++n) {
            const float v = ldsF[tid * NFILT + n];
            s += v; s2 += v * v;
        }
        const float m   = s  * (1.0f / NFILT);
        const float var = fmaxf(s2 * (1.0f / NFILT) - m * m, 0.0f);
        ldsM[tid] = m;
        ldsS[tid] = __frsqrt_rn(var);
    }
    __syncthreads();

    // ---- Phase 4: normalize + store ----
    for (int idx = tid; idx < MROWS * NFILT; idx += 256) {
        const int f = idx >> 6;
        const int n = idx & 63;
        const int frame = f0 + f;
        if (frame < NUM_FRAMES)
            out[frame * NFILT + n] = (ldsF[idx] - ldsM[f]) * ldsS[f];
    }
}

// ---------------------------------------------------------------------------
extern "C" void kernel_launch(void* const* d_in, const int* in_sizes, int n_in,
                              void* d_out, int out_size, void* d_ws, size_t ws_size,
                              hipStream_t stream) {
    const float* x       = (const float*)d_in[0];   // (1, 8400240) f32
    const float* filters = (const float*)d_in[1];   // (257, 64) f32
    float*       out     = (float*)d_out;           // (1, 50000, 64, 1) f32

    _Float16* Bt = (_Float16*)d_ws;                 // NPAD*KPAD f16 (439,296 B)
    _Float16* Ft = Bt + (size_t)NPAD * KPAD;        // NFILT*MELK f16 (36,864 B)

    build_basis_kernel<<<(NPAD * KPAD + 255) / 256, 256, 0, stream>>>(Bt);
    build_filt_kernel <<<(NFILT * MELK + 255) / 256, 256, 0, stream>>>(filters, Ft);
    const int nblocks = (NUM_FRAMES + MROWS - 1) / MROWS;   // 1563
    fbank_main_kernel<<<nblocks, 256, 0, stream>>>(x, Bt, Ft, out);
    (void)in_sizes; (void)n_in; (void)out_size; (void)ws_size;
}
